// DyGFormer_73701638799824
// MI455X (gfx1250) — matrix operand order repaired and checked by reading, not compiled, not run on hardware
//
#include <hip/hip_runtime.h>

#define BDIM 256
#define NID_ 10000
#define L_ 512
#define D_ 64
#define B_ 256

typedef __attribute__((ext_vector_type(16))) _Float16 v16h;
typedef __attribute__((ext_vector_type(2)))  _Float16 v2h;
typedef __attribute__((ext_vector_type(8)))  float    v8f;

__global__ __launch_bounds__(BDIM) void dygformer_fused_kernel(
    const int* __restrict__ src_ids, const int* __restrict__ dst_ids,
    const float* __restrict__ W1, const float* __restrict__ b1,
    const float* __restrict__ W2, const float* __restrict__ b2,
    float* __restrict__ out_src, float* __restrict__ out_dst)
{
  // Packed per-sample histogram: lo16 = count in src seq, hi16 = count in dst seq.
  __shared__ unsigned int cnt[NID_];
  __shared__ _Float16 sW1h[D_], sb1h[D_];
  __shared__ float sb2x2[D_];

  const int b   = blockIdx.x;
  const int tid = threadIdx.x;

  for (int i = tid; i < NID_; i += BDIM) cnt[i] = 0u;
  if (tid < D_) {
    sW1h[tid]  = (_Float16)W1[tid];   // W1 is (D,1) -> flat D floats
    sb1h[tid]  = (_Float16)b1[tid];
    sb2x2[tid] = 2.0f * b2[tid];      // channel-sum doubles the bias
  }
  __syncthreads();

  for (int i = tid; i < L_; i += BDIM) {
    atomicAdd(&cnt[src_ids[b * L_ + i]], 1u);
    atomicAdd(&cnt[dst_ids[b * L_ + i]], 0x10000u);
  }
  __syncthreads();

  const int lane = tid & 31;
  const int wave = tid >> 5;
  const int n    = lane & 15;   // N (e) index within a 16-tile; also A-row m
  const int hi   = lane >> 4;   // half-wave select

  // ---- Load B fragments (W2 as K x N, B[d][e] = W2[e*D + d]) once per wave ----
  // B k order used here (contiguous, same as A, not AMD's table order): lanes 0-15 -> K = 0..15 (2 per VGPR), lanes 16-31 -> K = 16..31.
  v16h bf[4][2];
  {
    const int kbase_b = hi ? 16 : 0;
    #pragma unroll
    for (int t = 0; t < 4; ++t) {
      const int e = 16 * t + n;
      #pragma unroll
      for (int q = 0; q < 2; ++q) {
        #pragma unroll
        for (int j = 0; j < 8; ++j) {
          const int d0 = q * 32 + kbase_b + 2 * j;
          bf[t][q][2 * j]     = (_Float16)W2[e * D_ + d0];
          bf[t][q][2 * j + 1] = (_Float16)W2[e * D_ + d0 + 1];
        }
      }
    }
  }

  float cinit[4];
  #pragma unroll
  for (int t = 0; t < 4; ++t) cinit[t] = sb2x2[16 * t + n];

  // 16-bit A layout: lanes 0-15 -> K = 0..15, lanes 16-31 -> K = 16..31 (contiguous, matches B).
  const int kbase_a = hi ? 16 : 0;

  v2h zero2;
  zero2[0] = (_Float16)0.0f;
  zero2[1] = (_Float16)0.0f;

  // Per-lane base pointers; all per-store deltas are compile-time constants that
  // fit in the 24-bit instruction offset, so only these advance in the loop.
  const size_t lane_off0 = ((size_t)b * L_ + (size_t)(wave * 16 + 8 * hi)) * D_ + n;
  float* __restrict__ base_src = out_src + lane_off0;
  float* __restrict__ base_dst = out_dst + lane_off0;
  const size_t tile_step = (size_t)(BDIM / 32) * 16 * D_;  // 8 tiles * 16 rows * D

  const int* __restrict__ sidp = src_ids + b * L_ + wave * 16 + n;
  const int* __restrict__ didp = dst_ids + b * L_ + wave * 16 + n;

  // L/16 = 32 position tiles, 8 waves -> 4 tiles per wave.
  for (int tile = wave; tile < L_ / 16; tile += BDIM / 32) {
    const int sid = *sidp;
    const int did = *didp;
    const unsigned cs = cnt[sid];
    const unsigned cd = cnt[did];

    // Counts are <= 512: exactly representable in f16.
    float app[2][2];
    app[0][0] = (sid == 0) ? 0.0f : (float)(cs & 0xFFFFu);  // src in src
    app[0][1] = (sid == 0) ? 0.0f : (float)(cs >> 16);      // src in dst
    app[1][0] = (did == 0) ? 0.0f : (float)(cd & 0xFFFFu);  // dst in src
    app[1][1] = (did == 0) ? 0.0f : (float)(cd >> 16);      // dst in dst

    #pragma unroll
    for (int s = 0; s < 2; ++s) {
      v2h a1v, a2v;
      a1v[0] = a1v[1] = (_Float16)app[s][0];
      a2v[0] = a2v[1] = (_Float16)app[s][1];

      // A[m,d] = relu(a1*W1[d]+b1[d]) + relu(a2*W1[d]+b1[d]) in packed f16:
      // one pk_fma + one pk_max per channel, one pk_add per d-pair.
      v16h af[2];
      #pragma unroll
      for (int q = 0; q < 2; ++q) {
        #pragma unroll
        for (int j = 0; j < 8; ++j) {
          const int k0 = kbase_a + 2 * j;
          const int d  = q * 32 + k0;            // always even
          const v2h w  = *(const v2h*)(sW1h + d);
          const v2h bb = *(const v2h*)(sb1h + d);
          const v2h h  = __builtin_elementwise_max(a1v * w + bb, zero2)
                       + __builtin_elementwise_max(a2v * w + bb, zero2);
          af[q][2 * j]     = h[0];
          af[q][2 * j + 1] = h[1];
        }
      }

      float* __restrict__ base = s ? base_dst : base_src;
      #pragma unroll
      for (int t = 0; t < 4; ++t) {
        v8f c;
        #pragma unroll
        for (int i = 0; i < 8; ++i) c[i] = cinit[t];
        c = __builtin_amdgcn_wmma_f32_16x16x32_f16(false, af[0], false, bf[t][0],
                                                   (short)0, c, false, false);
        c = __builtin_amdgcn_wmma_f32_16x16x32_f16(false, af[1], false, bf[t][1],
                                                   (short)0, c, false, false);
        // D layout: c[i] is (m = i + 8*hi, e = 16t + n). Stores coalesce over n;
        // per-store deltas (i*D + 16*t floats) are immediate offsets.
        #pragma unroll
        for (int i = 0; i < 8; ++i) {
          base[(size_t)(i * D_ + 16 * t)] = c[i];
        }
      }
    }

    base_src += tile_step;
    base_dst += tile_step;
    sidp     += (BDIM / 32) * 16;
    didp     += (BDIM / 32) * 16;
  }
}

extern "C" void kernel_launch(void* const* d_in, const int* in_sizes, int n_in,
                              void* d_out, int out_size, void* d_ws, size_t ws_size,
                              hipStream_t stream) {
  (void)in_sizes; (void)n_in; (void)out_size; (void)d_ws; (void)ws_size;
  const int*   src_ids = (const int*)d_in[0];
  const int*   dst_ids = (const int*)d_in[1];
  const float* W1      = (const float*)d_in[2];
  const float* b1      = (const float*)d_in[3];
  const float* W2      = (const float*)d_in[4];
  const float* b2      = (const float*)d_in[5];
  float* out      = (float*)d_out;
  float* out_src  = out;
  float* out_dst  = out + (size_t)B_ * L_ * D_;

  dygformer_fused_kernel<<<B_, BDIM, 0, stream>>>(src_ids, dst_ids, W1, b1, W2, b2,
                                                  out_src, out_dst);
}
